// FlashMultiHeadSelfAttention_81157702025458
// MI455X (gfx1250) — matrix operand order repaired and checked by reading, not compiled, not run on hardware
//
#include <hip/hip_runtime.h>

typedef __attribute__((ext_vector_type(16))) __bf16       bf16x16;
typedef __attribute__((ext_vector_type(8)))  float        f32x8;
typedef __attribute__((ext_vector_type(8)))  unsigned int uintx8;
typedef __attribute__((ext_vector_type(4)))  unsigned int uintx4;
typedef __attribute__((ext_vector_type(8)))  int          intx8;
typedef __attribute__((ext_vector_type(4)))  int          intx4;

#define EMBED 1024
#define NHEAD 16
#define HDIM  64
#define BATCH 4
#define SEQ   2048
#define NTOK  (BATCH * SEQ)          // 8192
#define QKV_F (3 * EMBED)            // 3072

#if __has_builtin(__builtin_amdgcn_tensor_load_to_lds)
#define HAVE_TDM 1
#else
#define HAVE_TDM 0
#endif

// ---------------------------------------------------------------------------
// Tensor Data Mover: async 2D bf16 tile copy global -> LDS (D# per ISA 8.3/8.4)
// 6-arg builtin form: (g0, g1, g2, g3, g4, cpol)
// ---------------------------------------------------------------------------
#if HAVE_TDM
__device__ __forceinline__ void tdm_load_2d_bf16(unsigned lds_addr, const void* gptr,
                                                 unsigned tile_rows, unsigned tile_cols,
                                                 unsigned row_stride_elems,
                                                 unsigned tensor_rows) {
  unsigned long long ga = (unsigned long long)gptr;
  uintx4 g0;
  g0[0] = 1u;                                              // count=1, user mode
  g0[1] = lds_addr;                                        // LDS byte address
  g0[2] = (unsigned)(ga & 0xFFFFFFFFu);                    // global_addr[31:0]
  g0[3] = (unsigned)((ga >> 32) & 0x1FFFFFFu) | (2u << 30); // addr[56:32] | type=2
  intx8 g1;
  g1[0] = (int)(1u << 16);                 // data_size=1 (2 bytes/elem)
  g1[1] = (int)(tile_cols << 16);          // tensor_dim0[15:0]  (bits 63:48)
  g1[2] = (int)(tensor_rows << 16);        // dim0 hi=0 | tensor_dim1[15:0]
  g1[3] = (int)(tile_cols << 16);          // tile_dim0          (bits 127:112)
  g1[4] = (int)tile_rows;                  // tile_dim1          (bits 143:128)
  g1[5] = (int)row_stride_elems;           // tensor_dim0_stride[31:0]
  g1[6] = 0;
  g1[7] = 0;
  intx4 z4 = {0, 0, 0, 0};
  intx8 z8 = {0, 0, 0, 0, 0, 0, 0, 0};
  __builtin_amdgcn_tensor_load_to_lds(g0, g1, z4, z4, z8, 0);
}
#endif

// ---------------------------------------------------------------------------
// WMMA fragment helpers (wave32, 16x16x32 bf16 shapes, ISA 7.12.2 layouts)
// ---------------------------------------------------------------------------
__device__ __forceinline__ bf16x16 load_a_frag(const unsigned* base, int row_stride_u,
                                               int row, int lg) {
  uintx8 u;
#pragma unroll
  for (int g = 0; g < 8; ++g) {
    int k2 = (g < 4) ? (lg * 4 + g) : (8 + lg * 4 + (g - 4));   // k/2
    u[g] = base[row * row_stride_u + k2];
  }
  return __builtin_bit_cast(bf16x16, u);
}

__device__ __forceinline__ bf16x16 load_b_frag_nk(const unsigned* base, int row_stride_u,
                                                  int n, int lg, int k0) {
  uintx8 u;
#pragma unroll
  for (int g = 0; g < 8; ++g) {
    int k2 = (k0 >> 1) + ((g < 4) ? (lg * 4 + g) : (8 + lg * 4 + (g - 4)));   // k/2
    u[g] = base[n * row_stride_u + k2];
  }
  return __builtin_bit_cast(bf16x16, u);
}

__device__ __forceinline__ f32x8 wmma_bf16(bf16x16 a, bf16x16 b, f32x8 c) {
  return __builtin_amdgcn_wmma_f32_16x16x32_bf16(
      false, a, false, b, (short)0, c, false, false);
}

// Pack two f32 -> one dword of two bf16
__device__ __forceinline__ unsigned pack_bf16x2(float lo, float hi) {
  unsigned a = (unsigned)__builtin_bit_cast(unsigned short, (__bf16)lo);
  unsigned b = (unsigned)__builtin_bit_cast(unsigned short, (__bf16)hi);
  return a | (b << 16);
}

// ---------------------------------------------------------------------------
// Kernel 1: QKV = x @ W_qkv^T ; scatter into bf16 Q/K/V [B,H,S,D] buffers
// grid (NTOK/128, QKV_F/64), block 256 (8 waves: 4 in M x 2 in N, 32x32 each)
// ---------------------------------------------------------------------------
__global__ __launch_bounds__(256)
void qkv_gemm_kernel(const float* __restrict__ x, const float* __restrict__ Wqkv,
                     __bf16* __restrict__ Qb, __bf16* __restrict__ Kb,
                     __bf16* __restrict__ Vb) {
  __shared__ alignas(16) __bf16 As[128 * 32];
  __shared__ alignas(16) __bf16 Bs[64 * 32];

  const int tid  = threadIdx.x;
  const int lane = tid & 31, wave = tid >> 5;
  const int lg = lane >> 4, lr = lane & 15;
  const int mbase = blockIdx.x * 128;
  const int nbase = blockIdx.y * 64;
  const int wm = (wave & 3) * 32;
  const int wn = (wave >> 2) * 32;

  // Block-uniform output routing (one (c3,h,b) slice per block).
  const int c3 = nbase >> 10;
  const int h  = (nbase >> 6) & (NHEAD - 1);
  const int bb = mbase >> 11;
  const int sm = mbase & (SEQ - 1);
  __bf16* __restrict__ dst = (c3 == 0) ? Qb : (c3 == 1) ? Kb : Vb;
  const size_t obase = ((size_t)(bb * NHEAD + h) * SEQ + sm) * HDIM;

  const float2* x2 = (const float2*)x;
  const float2* w2 = (const float2*)Wqkv;
  unsigned* Asu = (unsigned*)As;
  unsigned* Bsu = (unsigned*)Bs;

  f32x8 acc[2][2] = {};

  for (int k0 = 0; k0 < EMBED; k0 += 32) {
#pragma unroll
    for (int i = 0; i < 8; ++i) {            // stage A: 128x32, pair-packed
      int p = tid + i * 256;                 // 2048 dword slots
      int r = p >> 4, cp = p & 15;
      float2 f = x2[(size_t)(mbase + r) * (EMBED / 2) + (k0 >> 1) + cp];
      Asu[p] = pack_bf16x2(f.x, f.y);
    }
#pragma unroll
    for (int i = 0; i < 4; ++i) {            // stage B: 64x32, pair-packed
      int p = tid + i * 256;
      int r = p >> 4, cp = p & 15;
      float2 f = w2[(size_t)(nbase + r) * (EMBED / 2) + (k0 >> 1) + cp];
      Bsu[p] = pack_bf16x2(f.x, f.y);
    }
    if (k0 + 32 < EMBED) {                   // prefetch next chunk
      __builtin_prefetch(&x[(size_t)(mbase + (tid >> 1)) * EMBED + k0 + 32], 0, 0);
      __builtin_prefetch(&Wqkv[(size_t)(nbase + (tid & 63)) * EMBED + k0 + 32], 0, 0);
    }
    __syncthreads();

    const unsigned* Au = (const unsigned*)As;
    const unsigned* Bu = (const unsigned*)Bs;
    bf16x16 a0 = load_a_frag(Au, 16, wm + lr, lg);
    bf16x16 a1 = load_a_frag(Au, 16, wm + 16 + lr, lg);
    bf16x16 b0 = load_b_frag_nk(Bu, 16, wn + lr, lg, 0);
    bf16x16 b1 = load_b_frag_nk(Bu, 16, wn + 16 + lr, lg, 0);

    acc[0][0] = wmma_bf16(a0, b0, acc[0][0]);
    acc[0][1] = wmma_bf16(a0, b1, acc[0][1]);
    acc[1][0] = wmma_bf16(a1, b0, acc[1][0]);
    acc[1][1] = wmma_bf16(a1, b1, acc[1][1]);
    __syncthreads();
  }

  // Branch-free scatter: row offset within slice + d column.
#pragma unroll
  for (int mt = 0; mt < 2; ++mt) {
#pragma unroll
    for (int nt = 0; nt < 2; ++nt) {
#pragma unroll
      for (int r = 0; r < 8; ++r) {
        int rowoff = wm + mt * 16 + lg * 8 + r;       // 0..127 within block
        int d      = wn + nt * 16 + lr;               // 0..63
        dst[obase + (size_t)rowoff * HDIM + d] = (__bf16)acc[mt][nt][r];
      }
    }
  }
}

// ---------------------------------------------------------------------------
// Kernel 2: flash attention; 64 query rows per 128-thread block
// K tile staged by the Tensor Data Mover; V tile transposed manually.
// grid (SEQ/64, BATCH*NHEAD)
// ---------------------------------------------------------------------------
__global__ __launch_bounds__(128)
void flash_attn_kernel(const __bf16* __restrict__ Qb, const __bf16* __restrict__ Kb,
                       const __bf16* __restrict__ Vb, __bf16* __restrict__ Of) {
  __shared__ alignas(16) __bf16 Kt[32 * 64];      // [key][d]
  __shared__ alignas(16) __bf16 Vt[64 * 32];      // [d][key]  (transposed)
  __shared__ alignas(16) __bf16 Ps[4][16 * 32];   // per-wave P tile [m][key]

  const int tid  = threadIdx.x;
  const int lane = tid & 31, wave = tid >> 5;
  const int lg = lane >> 4, lr = lane & 15;
  const int bh = blockIdx.y;
  const int b  = bh >> 4;
  const int h  = bh & (NHEAD - 1);
  const int qbase = blockIdx.x * 64 + wave * 16;

  // Q fragments: A-layout gather straight from global (uint = bf16 pair).
  const unsigned* Qu = (const unsigned*)Qb;
  const size_t qrow_u = ((size_t)bh * SEQ + qbase + lr) * (HDIM / 2);
  bf16x16 aq[2];
#pragma unroll
  for (int f = 0; f < 2; ++f) {
    uintx8 u;
#pragma unroll
    for (int g = 0; g < 8; ++g) {
      int k2 = f * 16 + ((g < 4) ? (lg * 4 + g) : (8 + lg * 4 + (g - 4)));
      u[g] = Qu[qrow_u + k2];
    }
    aq[f] = __builtin_bit_cast(bf16x16, u);
  }

  float mrow[8], lrow[8];
#pragma unroll
  for (int r = 0; r < 8; ++r) { mrow[r] = -1e30f; lrow[r] = 0.0f; }
  f32x8 o[4] = {};

  const float inv_sqrt_d = 0.125f;   // 1/sqrt(64)

  for (int kb = 0; kb < SEQ / 32; ++kb) {
    const __bf16* kblk = Kb + ((size_t)bh * SEQ + kb * 32) * HDIM;
#if HAVE_TDM
    if (wave == 0) {                        // TDM copies 32x64 bf16 row-major
      tdm_load_2d_bf16((unsigned)(uintptr_t)(void*)Kt, kblk,
                       /*rows=*/32, /*cols=*/64, /*stride=*/HDIM,
                       /*tensor_rows=*/SEQ);
    }
#else
    {
      const uint4* ksrc = (const uint4*)kblk;
      uint4* kdst = (uint4*)Kt;
      kdst[tid]       = ksrc[tid];
      kdst[tid + 128] = ksrc[tid + 128];
    }
#endif
    const __bf16* vsrc = Vb + ((size_t)bh * SEQ + kb * 32) * HDIM;
#pragma unroll
    for (int i = 0; i < 16; ++i) {
      int e = tid + i * 128;
      int key = e >> 6, d = e & 63;
      Vt[d * 32 + key] = vsrc[e];
    }
#if HAVE_TDM
    if (wave == 0) __builtin_amdgcn_s_wait_tensorcnt(0);
#endif
    __syncthreads();

    // Scores: S[16 x 32] = Q(16x64) . K^T, two k-steps of 32 over d.
    const unsigned* Ku = (const unsigned*)Kt;
    f32x8 s0 = {}, s1 = {};
#pragma unroll
    for (int f = 0; f < 2; ++f) {
      bf16x16 bk0 = load_b_frag_nk(Ku, 32, lr, lg, f * 32);
      bf16x16 bk1 = load_b_frag_nk(Ku, 32, 16 + lr, lg, f * 32);
      s0 = wmma_bf16(aq[f], bk0, s0);
      s1 = wmma_bf16(aq[f], bk1, s1);
    }

    // Online softmax per row (row m = lg*8 + r; N spread across 16-lane half).
    float resc[8];
#pragma unroll
    for (int r = 0; r < 8; ++r) {
      float v0 = s0[r] * inv_sqrt_d;
      float v1 = s1[r] * inv_sqrt_d;
      float mx = fmaxf(v0, v1);
#pragma unroll
      for (int msk = 1; msk <= 8; msk <<= 1) mx = fmaxf(mx, __shfl_xor(mx, msk, 32));
      float mnew = fmaxf(mrow[r], mx);
      float rs   = __expf(mrow[r] - mnew);
      float p0   = __expf(v0 - mnew);
      float p1   = __expf(v1 - mnew);
      float sm   = p0 + p1;
#pragma unroll
      for (int msk = 1; msk <= 8; msk <<= 1) sm += __shfl_xor(sm, msk, 32);
      lrow[r] = lrow[r] * rs + sm;
      mrow[r] = mnew;
      resc[r] = rs;
      int m = lg * 8 + r;
      Ps[wave][m * 32 + lr]      = (__bf16)p0;
      Ps[wave][m * 32 + 16 + lr] = (__bf16)p1;
    }

#pragma unroll
    for (int dt = 0; dt < 4; ++dt)
#pragma unroll
      for (int r = 0; r < 8; ++r) o[dt][r] *= resc[r];

    const unsigned* Pu = (const unsigned*)Ps[wave];
    bf16x16 ap = load_a_frag(Pu, 16, lr, lg);
    const unsigned* Vu = (const unsigned*)Vt;
#pragma unroll
    for (int dt = 0; dt < 4; ++dt) {
      bf16x16 bv = load_b_frag_nk(Vu, 16, dt * 16 + lr, lg, 0);  // [d][key]
      o[dt] = wmma_bf16(ap, bv, o[dt]);
    }
    __syncthreads();
  }

  float invl[8];
#pragma unroll
  for (int r = 0; r < 8; ++r) invl[r] = 1.0f / lrow[r];
#pragma unroll
  for (int dt = 0; dt < 4; ++dt) {
#pragma unroll
    for (int r = 0; r < 8; ++r) {
      int q = qbase + lg * 8 + r;
      int d = dt * 16 + lr;
      size_t idx = ((size_t)(b * SEQ + q)) * EMBED + h * HDIM + d;
      Of[idx] = (__bf16)(o[dt][r] * invl[r]);
    }
  }
}

// ---------------------------------------------------------------------------
// Kernel 3: out = O_flat(bf16) @ W_out^T -> f32
// A tile staged by TDM; B tile converted f32->bf16 by the waves.
// grid (NTOK/128, EMBED/64), block 256
// ---------------------------------------------------------------------------
__global__ __launch_bounds__(256)
void out_gemm_kernel(const __bf16* __restrict__ Of, const float* __restrict__ Wout,
                     float* __restrict__ out) {
  __shared__ alignas(16) __bf16 As[128 * 32];
  __shared__ alignas(16) __bf16 Bs[64 * 32];

  const int tid  = threadIdx.x;
  const int lane = tid & 31, wave = tid >> 5;
  const int lg = lane >> 4, lr = lane & 15;
  const int mbase = blockIdx.x * 128;
  const int nbase = blockIdx.y * 64;
  const int wm = (wave & 3) * 32;
  const int wn = (wave >> 2) * 32;

  const float2* w2 = (const float2*)Wout;
  unsigned* Bsu = (unsigned*)Bs;

  f32x8 acc[2][2] = {};

  for (int k0 = 0; k0 < EMBED; k0 += 32) {
#if HAVE_TDM
    if (wave == 0) {                        // TDM copies 128x32 bf16 tile
      tdm_load_2d_bf16((unsigned)(uintptr_t)(void*)As,
                       Of + (size_t)mbase * EMBED + k0,
                       /*rows=*/128, /*cols=*/32, /*stride=*/EMBED,
                       /*tensor_rows=*/NTOK);
    }
#else
    {
      const unsigned* srcu = (const unsigned*)Of;
#pragma unroll
      for (int i = 0; i < 8; ++i) {
        int e = tid + i * 256;
        int r = e >> 4, cu = e & 15;
        ((unsigned*)As)[e] = srcu[(size_t)(mbase + r) * (EMBED / 2) + (k0 >> 1) + cu];
      }
    }
#endif
#pragma unroll
    for (int i = 0; i < 4; ++i) {           // stage B: 64x32 f32 -> bf16 packed
      int p = tid + i * 256;
      int r = p >> 4, cp = p & 15;
      float2 f = w2[(size_t)(nbase + r) * (EMBED / 2) + (k0 >> 1) + cp];
      Bsu[p] = pack_bf16x2(f.x, f.y);
    }
    if (k0 + 32 < EMBED)
      __builtin_prefetch(&Wout[(size_t)(nbase + (tid & 63)) * EMBED + k0 + 32], 0, 0);
#if HAVE_TDM
    if (wave == 0) __builtin_amdgcn_s_wait_tensorcnt(0);
#endif
    __syncthreads();

    const unsigned* Au = (const unsigned*)As;
    const unsigned* Bu = (const unsigned*)Bs;
    bf16x16 a0 = load_a_frag(Au, 16, wm + lr, lg);
    bf16x16 a1 = load_a_frag(Au, 16, wm + 16 + lr, lg);
    bf16x16 b0 = load_b_frag_nk(Bu, 16, wn + lr, lg, 0);
    bf16x16 b1 = load_b_frag_nk(Bu, 16, wn + 16 + lr, lg, 0);

    acc[0][0] = wmma_bf16(a0, b0, acc[0][0]);
    acc[0][1] = wmma_bf16(a0, b1, acc[0][1]);
    acc[1][0] = wmma_bf16(a1, b0, acc[1][0]);
    acc[1][1] = wmma_bf16(a1, b1, acc[1][1]);
    __syncthreads();
  }

#pragma unroll
  for (int mt = 0; mt < 2; ++mt) {
#pragma unroll
    for (int nt = 0; nt < 2; ++nt) {
#pragma unroll
      for (int r = 0; r < 8; ++r) {
        int t = mbase + wm + mt * 16 + lg * 8 + r;
        int f = nbase + wn + nt * 16 + lr;
        out[(size_t)t * EMBED + f] = acc[mt][nt][r];
      }
    }
  }
}

// ---------------------------------------------------------------------------
// Launch
// ---------------------------------------------------------------------------
extern "C" void kernel_launch(void* const* d_in, const int* in_sizes, int n_in,
                              void* d_out, int out_size, void* d_ws, size_t ws_size,
                              hipStream_t stream) {
  (void)in_sizes; (void)n_in; (void)out_size; (void)ws_size;

  const float* x    = (const float*)d_in[0];   // [4, 2048, 1024]
  const float* Wqkv = (const float*)d_in[1];   // [3072, 1024]
  const float* Wout = (const float*)d_in[2];   // [1024, 1024]
  float*       out  = (float*)d_out;           // [4, 2048, 1024]

  const size_t elems = (size_t)NTOK * EMBED;   // 8M elements per buffer
  __bf16* Qb = (__bf16*)d_ws;
  __bf16* Kb = Qb + elems;
  __bf16* Vb = Kb + elems;
  __bf16* Of = Vb + elems;                     // total 64 MB of ws

  qkv_gemm_kernel<<<dim3(NTOK / 128, QKV_F / 64), 256, 0, stream>>>(x, Wqkv, Qb, Kb, Vb);
  flash_attn_kernel<<<dim3(SEQ / 64, BATCH * NHEAD), 128, 0, stream>>>(Qb, Kb, Vb, Of);
  out_gemm_kernel<<<dim3(NTOK / 128, EMBED / 64), 256, 0, stream>>>(Of, Wout, out);
}
